// SelfAttentionGen_53360673685803
// MI455X (gfx1250) — matrix operand order repaired and checked by reading, not compiled, not run on hardware
//
#include <hip/hip_runtime.h>

typedef __attribute__((ext_vector_type(16))) _Float16 v16h;
typedef __attribute__((ext_vector_type(8)))  _Float16 v8h;
typedef __attribute__((ext_vector_type(8)))  float    v8f;
typedef __attribute__((ext_vector_type(4)))  int      v4i;

#define TT 2048
#define KDIM 1024
#define NHEAD 16
#define HSZ 64
#define MROWS 4096  // b*t

// LDS row strides (halfs): 16B-aligned rows, non-power-of-two word stride
#define LDK 72   // K tile rows (32 keys x 64 d)
#define LDV 40   // V tile rows (64 d x 32 keys)
#define LDP 40   // P tile rows (16 q x 32 keys)

// ---------- CDNA5 async global->LDS copy (guarded; fallback = sync relay) ----
#if __has_builtin(__builtin_amdgcn_global_load_async_to_lds_b128)
#define HAVE_ASYNC_LDS 1
typedef __attribute__((address_space(1))) v4i gv4i;  // global int4
typedef __attribute__((address_space(3))) v4i sv4i;  // LDS int4
__device__ __forceinline__ void async_cp16(_Float16* dst, const _Float16* src) {
  __builtin_amdgcn_global_load_async_to_lds_b128((gv4i*)src, (sv4i*)dst, 0, 0);
}
__device__ __forceinline__ void wait_async() {
  asm volatile("s_wait_asynccnt 0x0" ::: "memory");
}
#else
#define HAVE_ASYNC_LDS 0
__device__ __forceinline__ void async_cp16(_Float16* dst, const _Float16* src) {
  *(v8h*)dst = *(const v8h*)src;
}
__device__ __forceinline__ void wait_async() {}
#endif

// ---------- WMMA fragment loaders (ISA 7.12.2 layouts, wave32) ----------

// A-matrix 16x32 f16: lane l (l<16) holds row l; hi=lane>>4 selects K-halves:
//   comps 0..7  = K[8*hi .. +7],  comps 8..15 = K[16+8*hi .. +7]
__device__ __forceinline__ v16h load_a_frag(const _Float16* __restrict__ src,
                                            int row0, int k0, int ld) {
  int lane = threadIdx.x & 31;
  int l = lane & 15, hi = lane >> 4;
  const _Float16* p = src + (size_t)(row0 + l) * ld + k0 + 8 * hi;
  v8h lo = *(const v8h*)(p);
  v8h hv = *(const v8h*)(p + 16);
  return __builtin_shufflevector(lo, hv, 0,1,2,3,4,5,6,7,8,9,10,11,12,13,14,15);
}

// B-matrix 32x16 f16 (K x N): lane l holds column col0+l; comps 0..7 = K[8*hi..+7],
// comps 8..15 = K[16+8*hi..+7]. src is row-major [N, K] (each B column == one contiguous src row).
__device__ __forceinline__ v16h load_b_frag(const _Float16* __restrict__ src,
                                            int col0, int k0, int ld) {
  int lane = threadIdx.x & 31;
  int l = lane & 15, hi = lane >> 4;
  const _Float16* p = src + (size_t)(col0 + l) * ld + k0 + 8 * hi;
  v8h lo = *(const v8h*)(p);
  v8h hv = *(const v8h*)(p + 16);
  return __builtin_shufflevector(lo, hv, 0,1,2,3,4,5,6,7,8,9,10,11,12,13,14,15);
}

__device__ __forceinline__ v8f wmma_f16(v16h a, v16h b, v8f c) {
  return __builtin_amdgcn_wmma_f32_16x16x32_f16(false, a, false, b, (short)0, c,
                                                false, false);
}

// ---------- f32 -> f16 convert ----------
__global__ void f32_to_f16_kernel(const float* __restrict__ src,
                                  _Float16* __restrict__ dst, int n) {
  int i = blockIdx.x * blockDim.x + threadIdx.x;
  if (i < n) dst[i] = (_Float16)src[i];
}

// ---------- GEMM: Y[i,j] = sum_d A[i,d] * W[j,d]  (A:[M,K], W:[N,K], f16) ----
// mode 0: f16 store, head layout    [b,h,t,hs]   (Q, K projections)
// mode 2: f16 store, head-transposed [b,h,hs,t]  (V projection)
// mode 1: f32 store + bias, row-major [M,N]      (output projection)
__global__ void __launch_bounds__(256)
gemm_xwT_kernel(const _Float16* __restrict__ A, const _Float16* __restrict__ W,
                _Float16* __restrict__ out_h, float* __restrict__ out_f,
                const float* __restrict__ bias, int Kd, int mode) {
  int wave = threadIdx.x >> 5;
  int lane = threadIdx.x & 31;
  int l = lane & 15, hi = lane >> 4;
  int row0 = blockIdx.y * 128 + (wave >> 2) * 64;   // 2 waves along M (64 rows)
  int col0 = blockIdx.x * 128 + (wave & 3) * 32;    // 4 waves along N (32 cols)

  v8f acc[4][2] = {};
  for (int k0 = 0; k0 < Kd; k0 += 32) {
    v16h af[4], bf[2];
#pragma unroll
    for (int mi = 0; mi < 4; ++mi) af[mi] = load_a_frag(A, row0 + 16 * mi, k0, Kd);
#pragma unroll
    for (int ni = 0; ni < 2; ++ni) bf[ni] = load_b_frag(W, col0 + 16 * ni, k0, Kd);
#pragma unroll
    for (int mi = 0; mi < 4; ++mi)
#pragma unroll
      for (int ni = 0; ni < 2; ++ni)
        acc[mi][ni] = wmma_f16(af[mi], bf[ni], acc[mi][ni]);
  }

#pragma unroll
  for (int mi = 0; mi < 4; ++mi) {
#pragma unroll
    for (int ni = 0; ni < 2; ++ni) {
#pragma unroll
      for (int r = 0; r < 8; ++r) {
        int grow = row0 + 16 * mi + r + 8 * hi;   // C layout: v[r] -> M=r / M=r+8
        int gcol = col0 + 16 * ni + l;
        float v = acc[mi][ni][r];
        if (mode == 1) {
          out_f[(size_t)grow * KDIM + gcol] = v + bias[gcol];
        } else {
          int b_ = grow / TT, tt = grow % TT;
          int hd = gcol / HSZ, d = gcol % HSZ;
          if (mode == 0)  // [b,h,t,hs]
            out_h[(((size_t)b_ * NHEAD + hd) * TT + tt) * HSZ + d] = (_Float16)v;
          else            // [b,h,hs,t]
            out_h[(((size_t)b_ * NHEAD + hd) * HSZ + d) * TT + tt] = (_Float16)v;
        }
      }
    }
  }
}

// ---------- fused causal flash attention, block-staged K/V ----------
// Q,K: [b,h,t,hs] f16; Vt: [b,h,hs,t] f16; Oh out: [b,t,h*hs] f16
// grid: (t/128, b*h), 256 threads = 8 waves; each wave owns 16 query rows.
// K/V tiles (32 keys) are async-staged into double-buffered LDS by the whole
// block; the j-loop trip count is block-uniform so barriers are legal, and
// waves past their causal bound skip only the compute body.
__global__ void __launch_bounds__(256)
flash_attn_kernel(const _Float16* __restrict__ Q, const _Float16* __restrict__ Km,
                  const _Float16* __restrict__ Vt, const int* __restrict__ padding,
                  _Float16* __restrict__ Oh) {
  __shared__ __align__(16) _Float16 ldsK[2][32 * LDK];  // [key][d]
  __shared__ __align__(16) _Float16 ldsV[2][64 * LDV];  // [d][key]
  __shared__ __align__(16) _Float16 plds[8][16 * LDP];  // per-wave P tile

  int tid = threadIdx.x;
  int wave = tid >> 5, lane = tid & 31;
  int l = lane & 15, hi = lane >> 4;
  int bh = blockIdx.y, b_ = bh >> 4, h_ = bh & 15;
  int q0blk = blockIdx.x * 128;
  int q0 = q0blk + wave * 16;

  const _Float16* Qb = Q + (size_t)bh * TT * HSZ;
  const _Float16* Kb = Km + (size_t)bh * TT * HSZ;
  const _Float16* Vb = Vt + (size_t)bh * HSZ * TT;
  const int* pad = padding + (size_t)b_ * TT;

  // Q fragments for this wave's 16 rows, resident for the whole pass
  v16h aq0 = load_a_frag(Qb, q0, 0, HSZ);
  v16h aq1 = load_a_frag(Qb, q0, 32, HSZ);

  float mrow[8], lrow[8];
#pragma unroll
  for (int r = 0; r < 8; ++r) { mrow[r] = -3.0e38f; lrow[r] = 0.0f; }
  v8f accO[4] = {};

  // per-thread staging coordinates: one b128 of K, one b128 of V each
  const int kr = tid >> 3, kc = (tid & 7) * 8;  // K: key row, d chunk
  const int vd = tid >> 2, vc = (tid & 3) * 8;  // V: d row, key chunk

  // prologue: stage tile 0 into buffer 0
  async_cp16(&ldsK[0][kr * LDK + kc], Kb + (size_t)kr * HSZ + kc);
  async_cp16(&ldsV[0][vd * LDV + vc], Vb + (size_t)vd * TT + vc);

  const float scale = 0.125f;  // 1/sqrt(64)
  const int blockJmax = q0blk + 127;  // block-uniform causal bound
  const int qmax = q0 + 15;           // this wave's causal bound
  int it = 0;
  for (int j0 = 0; j0 <= blockJmax; j0 += 32, ++it) {
    int buf = it & 1;
    wait_async();        // my async copies into `buf` have landed
    __syncthreads();     // everyone's copies landed; prior reads of buf^1 done
    if (j0 + 32 <= blockJmax) {
      int jn = j0 + 32;
      async_cp16(&ldsK[buf ^ 1][kr * LDK + kc], Kb + (size_t)(jn + kr) * HSZ + kc);
      async_cp16(&ldsV[buf ^ 1][vd * LDV + vc], Vb + (size_t)vd * TT + jn + vc);
    }
    if (j0 > qmax) continue;  // fully-masked tile for this wave (wave-uniform)

    const _Float16* Kt = &ldsK[buf][0];
    const _Float16* Vtile = &ldsV[buf][0];

    // S = Q @ K^T : two 16x16 score tiles (keys j0.. and j0+16..)
    v8f s0 = {}, s1 = {};
    s0 = wmma_f16(aq0, load_b_frag(Kt, 0, 0, LDK), s0);
    s0 = wmma_f16(aq1, load_b_frag(Kt, 0, 32, LDK), s0);
    s1 = wmma_f16(aq0, load_b_frag(Kt, 16, 0, LDK), s1);
    s1 = wmma_f16(aq1, load_b_frag(Kt, 16, 32, LDK), s1);

    int key0 = j0 + l, key1 = j0 + 16 + l;
    bool p0mask = (pad[key0] == 0);
    bool p1mask = (pad[key1] == 0);

#pragma unroll
    for (int r = 0; r < 8; ++r) {
      int qrow = q0 + r + 8 * hi;  // C layout row for this lane-half
      float v0 = (p0mask || key0 > qrow) ? -3.0e38f : s0[r] * scale;
      float v1 = (p1mask || key1 > qrow) ? -3.0e38f : s1[r] * scale;
      float rm = fmaxf(v0, v1);
#pragma unroll
      for (int off = 8; off >= 1; off >>= 1) rm = fmaxf(rm, __shfl_xor(rm, off, 16));
      float mn = fmaxf(mrow[r], rm);
      float alpha = __expf(mrow[r] - mn);
      float e0 = __expf(v0 - mn);
      float e1 = __expf(v1 - mn);
      float rs = e0 + e1;
#pragma unroll
      for (int off = 8; off >= 1; off >>= 1) rs += __shfl_xor(rs, off, 16);
      lrow[r] = lrow[r] * alpha + rs;
      mrow[r] = mn;
#pragma unroll
      for (int nt = 0; nt < 4; ++nt) accO[nt][r] *= alpha;
      // stash P (f16) in per-wave LDS for C-layout -> A-layout lane transpose
      plds[wave][(r + 8 * hi) * LDP + l] = (_Float16)e0;
      plds[wave][(r + 8 * hi) * LDP + 16 + l] = (_Float16)e1;
    }
    asm volatile("" ::: "memory");  // order DS store -> DS load (in-order per wave)
    const _Float16* pp = &plds[wave][l * LDP + 8 * hi];
    v8h plo = *(const v8h*)(pp);
    v8h phi = *(const v8h*)(pp + 16);
    v16h pa = __builtin_shufflevector(plo, phi, 0,1,2,3,4,5,6,7,8,9,10,11,12,13,14,15);
    asm volatile("" ::: "memory");
    // accO += P @ V  (V tile is [d][key] so B columns are contiguous LDS rows)
#pragma unroll
    for (int nt = 0; nt < 4; ++nt)
      accO[nt] = wmma_f16(pa, load_b_frag(Vtile, 16 * nt, 0, LDV), accO[nt]);
  }

  // normalize and store to [b, t, h*hs]
#pragma unroll
  for (int nt = 0; nt < 4; ++nt) {
#pragma unroll
    for (int r = 0; r < 8; ++r) {
      int qrow = q0 + r + 8 * hi;
      int col = h_ * HSZ + 16 * nt + l;
      float v = accO[nt][r] / lrow[r];
      Oh[((size_t)b_ * TT + qrow) * KDIM + col] = (_Float16)v;
    }
  }
}

// ---------- launcher ----------
extern "C" void kernel_launch(void* const* d_in, const int* in_sizes, int n_in,
                              void* d_out, int out_size, void* d_ws, size_t ws_size,
                              hipStream_t stream) {
  (void)in_sizes; (void)n_in; (void)out_size; (void)ws_size;
  const float* x  = (const float*)d_in[0];   // [2,2048,1024]
  const int* padding = (const int*)d_in[1];  // [2,2048]
  const float* Wq = (const float*)d_in[2];
  const float* Wk = (const float*)d_in[3];
  const float* Wv = (const float*)d_in[4];
  const float* Wu = (const float*)d_in[5];
  const float* bu = (const float*)d_in[6];

  _Float16* ws = (_Float16*)d_ws;
  const size_t NX = (size_t)MROWS * KDIM;   // 4M
  const size_t NW = (size_t)KDIM * KDIM;    // 1M
  _Float16* xh  = ws;                 // 4M halfs
  _Float16* Wqh = xh + NX;            // 1M
  _Float16* Wkh = Wqh + NW;
  _Float16* Wvh = Wkh + NW;
  _Float16* Wuh = Wvh + NW;
  _Float16* Qb  = Wuh + NW;           // [b,h,t,hs] 4M
  _Float16* Kb  = Qb + NX;            // 4M
  _Float16* Vt  = Kb + NX;            // [b,h,hs,t] 4M
  _Float16* Oh  = xh;                 // reuse x-half buffer after projections

  int nx = (int)NX, nw = (int)NW;
  f32_to_f16_kernel<<<(nx + 255) / 256, 256, 0, stream>>>(x,  xh,  nx);
  f32_to_f16_kernel<<<(nw + 255) / 256, 256, 0, stream>>>(Wq, Wqh, nw);
  f32_to_f16_kernel<<<(nw + 255) / 256, 256, 0, stream>>>(Wk, Wkh, nw);
  f32_to_f16_kernel<<<(nw + 255) / 256, 256, 0, stream>>>(Wv, Wvh, nw);
  f32_to_f16_kernel<<<(nw + 255) / 256, 256, 0, stream>>>(Wu, Wuh, nw);

  dim3 gg(KDIM / 128, MROWS / 128);  // (8, 32)
  gemm_xwT_kernel<<<gg, 256, 0, stream>>>(xh, Wqh, Qb, nullptr, nullptr, KDIM, 0);
  gemm_xwT_kernel<<<gg, 256, 0, stream>>>(xh, Wkh, Kb, nullptr, nullptr, KDIM, 0);
  gemm_xwT_kernel<<<gg, 256, 0, stream>>>(xh, Wvh, Vt, nullptr, nullptr, KDIM, 2);

  flash_attn_kernel<<<dim3(TT / 128, 2 * NHEAD), 256, 0, stream>>>(Qb, Kb, Vt,
                                                                   padding, Oh);

  gemm_xwT_kernel<<<gg, 256, 0, stream>>>(Oh, Wuh, nullptr, (float*)d_out, bu,
                                          KDIM, 1);
}
